// MultiheadAttentionMY_50139448213811
// MI455X (gfx1250) — compile-verified, matrix operand order checked by reading, not run on hardware
//
#include <hip/hip_runtime.h>
#include <hip/hip_bf16.h>

typedef __attribute__((ext_vector_type(16))) _Float16 v16h;
typedef __attribute__((ext_vector_type(8)))  _Float16 v8h;
typedef __attribute__((ext_vector_type(8)))  float    v8f;

#define Bsz   8
#define Ssz   1024
#define IND   512
#define ORIG  64
#define EMB   512
#define NH    8
#define HD    64
#define KDIM  (IND + ORIG)      // 576
#define N_QKV (3 * EMB)         // 1536

__device__ __forceinline__ v8f wmma16(v16h a, v16h b, v8f c) {
    return __builtin_amdgcn_wmma_f32_16x16x32_f16(false, a, false, b, (short)0, c,
                                                  false, false);
}

union frag_u { v16h v; v8h h[2]; };

// Fragment from row-major f16 matrix; p points at (row, k0+kofs).
// Lane holds K = +0..7 and K = +16..23 relative to p (two b128 loads).
__device__ __forceinline__ v16h frag_f16(const _Float16* p) {
    frag_u u;
    u.h[0] = *(const v8h*)p;
    u.h[1] = *(const v8h*)(p + 16);
    return u.v;
}

// Same, source f32, converted on load (used only where data is read once).
__device__ __forceinline__ v16h frag_f32(const float* p) {
    v8f lo = *(const v8f*)p;
    v8f hi = *(const v8f*)(p + 16);
    v16h f;
#pragma unroll
    for (int i = 0; i < 8; ++i) { f[i] = (_Float16)lo[i]; f[8 + i] = (_Float16)hi[i]; }
    return f;
}

__device__ __forceinline__ v8h cvt8(v8f s) {
    v8h d;
#pragma unroll
    for (int j = 0; j < 8; ++j) d[j] = (_Float16)s[j];
    return d;
}

// ---------------------------------------------------------------------------
// P0a: generic f32 -> f16 (8 elems/thread).  grid = n8/256, block = 256
// ---------------------------------------------------------------------------
__global__ __launch_bounds__(256) void k_cvt16(
    const float* __restrict__ src, _Float16* __restrict__ dst, int n8)
{
    const int i = blockIdx.x * 256 + threadIdx.x;
    if (i < n8)
        *(v8h*)(dst + (size_t)i * 8) = cvt8(*(const v8f*)(src + (size_t)i * 8));
}

// ---------------------------------------------------------------------------
// P0b: xc = concat(x, inp) in f16, [8192, 576].  grid = 8192, block = 128
// ---------------------------------------------------------------------------
__global__ __launch_bounds__(128) void k_concat16(
    const float* __restrict__ x, const float* __restrict__ inp,
    _Float16* __restrict__ xc)
{
    const size_t g = blockIdx.x;
    const int t = threadIdx.x;
    if (t < 64) {
        *(v8h*)(xc + g * KDIM + t * 8) = cvt8(*(const v8f*)(x + g * IND + t * 8));
    } else if (t < 72) {
        const int u = t - 64;
        *(v8h*)(xc + g * KDIM + IND + u * 8) =
            cvt8(*(const v8f*)(inp + g * ORIG + u * 8));
    }
}

// ---------------------------------------------------------------------------
// K1: qkv = xc @ Wqkv^T + b  -> scatter to Q, K ([S,64]/head) and V^T
// ([64,S]/head), f16.  Wave tile 32x64.  grid = (8192/32, 1536/64), block = 32
// ---------------------------------------------------------------------------
__global__ __launch_bounds__(32) void k_qkv_gemm(
    const _Float16* __restrict__ xc, const _Float16* __restrict__ Wq,
    const float* __restrict__ bqkv,
    _Float16* __restrict__ qws, _Float16* __restrict__ kws,
    _Float16* __restrict__ vtws)
{
    const int lane = threadIdx.x;
    const int m0 = blockIdx.x * 32;
    const int n0 = blockIdx.y * 64;
    const int row  = lane & 15;
    const int kofs = (lane & 16) ? 8 : 0;

    v8f acc[2][4] = {};
    for (int kb = 0; kb < KDIM / 32; ++kb) {
        const int k0 = kb * 32;
        v16h a0 = frag_f16(xc + (size_t)(m0 + row) * KDIM + k0 + kofs);
        v16h a1 = frag_f16(xc + (size_t)(m0 + 16 + row) * KDIM + k0 + kofs);
        v16h b0 = frag_f16(Wq + (size_t)(n0 + row) * KDIM + k0 + kofs);
        v16h b1 = frag_f16(Wq + (size_t)(n0 + 16 + row) * KDIM + k0 + kofs);
        v16h b2 = frag_f16(Wq + (size_t)(n0 + 32 + row) * KDIM + k0 + kofs);
        v16h b3 = frag_f16(Wq + (size_t)(n0 + 48 + row) * KDIM + k0 + kofs);
        acc[0][0] = wmma16(a0, b0, acc[0][0]);
        acc[0][1] = wmma16(a0, b1, acc[0][1]);
        acc[0][2] = wmma16(a0, b2, acc[0][2]);
        acc[0][3] = wmma16(a0, b3, acc[0][3]);
        acc[1][0] = wmma16(a1, b0, acc[1][0]);
        acc[1][1] = wmma16(a1, b1, acc[1][1]);
        acc[1][2] = wmma16(a1, b2, acc[1][2]);
        acc[1][3] = wmma16(a1, b3, acc[1][3]);
    }

    const int col = lane & 15;
    const int mb  = (lane & 16) ? 8 : 0;
#pragma unroll
    for (int i = 0; i < 2; ++i)
#pragma unroll
        for (int t = 0; t < 4; ++t)
#pragma unroll
            for (int r = 0; r < 8; ++r) {
                const int n  = n0 + t * 16 + col;
                const int gm = m0 + i * 16 + mb + r;
                const int b2 = gm >> 10, s2 = gm & 1023;
                const float v = acc[i][t][r] + bqkv[n];
                const int h  = n / (3 * HD);
                const int rr = n - h * (3 * HD);
                const size_t bh = (size_t)b2 * NH + h;
                const _Float16 hv = (_Float16)v;
                if (rr < HD)          qws [(bh * Ssz + s2) * HD + rr]            = hv;
                else if (rr < 2 * HD) kws [(bh * Ssz + s2) * HD + (rr - HD)]     = hv;
                else                  vtws[(bh * HD + (rr - 2 * HD)) * Ssz + s2] = hv;
            }
}

// ---------------------------------------------------------------------------
// K2: logits = (Q.K) * 0.125 -> f32 into d_out attention slice.
// Wave tile 32x64.  grid = (1024/32, 1024/64, B*H), block = 32
// ---------------------------------------------------------------------------
__global__ __launch_bounds__(32) void k_logits(
    const _Float16* __restrict__ qws, const _Float16* __restrict__ kws,
    float* __restrict__ attn)
{
    const int lane = threadIdx.x;
    const int q0 = blockIdx.x * 32;
    const int n0 = blockIdx.y * 64;
    const size_t bh = blockIdx.z;
    const _Float16* Q = qws + bh * (size_t)Ssz * HD;
    const _Float16* K = kws + bh * (size_t)Ssz * HD;
    const int row  = lane & 15;
    const int kofs = (lane & 16) ? 8 : 0;

    v8f acc[2][4] = {};
#pragma unroll
    for (int kb = 0; kb < HD / 32; ++kb) {
        const int k0 = kb * 32;
        v16h a0 = frag_f16(Q + (size_t)(q0 + row) * HD + k0 + kofs);
        v16h a1 = frag_f16(Q + (size_t)(q0 + 16 + row) * HD + k0 + kofs);
        v16h b0 = frag_f16(K + (size_t)(n0 + row) * HD + k0 + kofs);
        v16h b1 = frag_f16(K + (size_t)(n0 + 16 + row) * HD + k0 + kofs);
        v16h b2 = frag_f16(K + (size_t)(n0 + 32 + row) * HD + k0 + kofs);
        v16h b3 = frag_f16(K + (size_t)(n0 + 48 + row) * HD + k0 + kofs);
        acc[0][0] = wmma16(a0, b0, acc[0][0]);
        acc[0][1] = wmma16(a0, b1, acc[0][1]);
        acc[0][2] = wmma16(a0, b2, acc[0][2]);
        acc[0][3] = wmma16(a0, b3, acc[0][3]);
        acc[1][0] = wmma16(a1, b0, acc[1][0]);
        acc[1][1] = wmma16(a1, b1, acc[1][1]);
        acc[1][2] = wmma16(a1, b2, acc[1][2]);
        acc[1][3] = wmma16(a1, b3, acc[1][3]);
    }

    float* out = attn + bh * (size_t)Ssz * Ssz;
    const int col = lane & 15;
    const int mb  = (lane & 16) ? 8 : 0;
#pragma unroll
    for (int i = 0; i < 2; ++i)
#pragma unroll
        for (int t = 0; t < 4; ++t)
#pragma unroll
            for (int r = 0; r < 8; ++r)
                out[(size_t)(q0 + i * 16 + mb + r) * Ssz + (n0 + t * 16 + col)] =
                    acc[i][t][r] * 0.125f;
}

// ---------------------------------------------------------------------------
// K3: in-place row softmax over 1024 cols. grid = B*H*S, block = 256
// ---------------------------------------------------------------------------
__global__ __launch_bounds__(256) void k_softmax(float* __restrict__ attn)
{
    __shared__ float red[256];
    float* p = attn + (size_t)blockIdx.x * Ssz;
    const int tid = threadIdx.x;

    float v[4];
    float m = -3.402823466e38f;
#pragma unroll
    for (int i = 0; i < 4; ++i) { v[i] = p[tid + i * 256]; m = fmaxf(m, v[i]); }
    red[tid] = m; __syncthreads();
    for (int s = 128; s > 0; s >>= 1) {
        if (tid < s) red[tid] = fmaxf(red[tid], red[tid + s]);
        __syncthreads();
    }
    m = red[0]; __syncthreads();

    float sum = 0.f;
#pragma unroll
    for (int i = 0; i < 4; ++i) { v[i] = __expf(v[i] - m); sum += v[i]; }
    red[tid] = sum; __syncthreads();
    for (int s = 128; s > 0; s >>= 1) {
        if (tid < s) red[tid] += red[tid + s];
        __syncthreads();
    }
    const float inv = 1.0f / red[0];
#pragma unroll
    for (int i = 0; i < 4; ++i) p[tid + i * 256] = v[i] * inv;
}

// ---------------------------------------------------------------------------
// K4: values = attn @ V  (A = f32 attn cvt on load — read exactly once;
// B from V^T).  Wave tile 32x64 (covers all d=64).
// grid = (1024/32, B*H), block = 32
// ---------------------------------------------------------------------------
__global__ __launch_bounds__(32) void k_attn_v(
    const float* __restrict__ attn, const _Float16* __restrict__ vtws,
    _Float16* __restrict__ val)
{
    const int lane = threadIdx.x;
    const int q0 = blockIdx.x * 32;
    const size_t bh = blockIdx.y;
    const int bb = (int)(bh >> 3), h = (int)(bh & 7);
    const float*    A  = attn + bh * (size_t)Ssz * Ssz;
    const _Float16* Vt = vtws + bh * (size_t)HD * Ssz;
    const int row  = lane & 15;
    const int kofs = (lane & 16) ? 8 : 0;

    v8f acc[2][4] = {};
    for (int kb = 0; kb < Ssz / 32; ++kb) {
        const int k0 = kb * 32;
        v16h a0 = frag_f32(A + (size_t)(q0 + row) * Ssz + k0 + kofs);
        v16h a1 = frag_f32(A + (size_t)(q0 + 16 + row) * Ssz + k0 + kofs);
        v16h b0 = frag_f16(Vt + (size_t)(row) * Ssz + k0 + kofs);
        v16h b1 = frag_f16(Vt + (size_t)(16 + row) * Ssz + k0 + kofs);
        v16h b2 = frag_f16(Vt + (size_t)(32 + row) * Ssz + k0 + kofs);
        v16h b3 = frag_f16(Vt + (size_t)(48 + row) * Ssz + k0 + kofs);
        acc[0][0] = wmma16(a0, b0, acc[0][0]);
        acc[0][1] = wmma16(a0, b1, acc[0][1]);
        acc[0][2] = wmma16(a0, b2, acc[0][2]);
        acc[0][3] = wmma16(a0, b3, acc[0][3]);
        acc[1][0] = wmma16(a1, b0, acc[1][0]);
        acc[1][1] = wmma16(a1, b1, acc[1][1]);
        acc[1][2] = wmma16(a1, b2, acc[1][2]);
        acc[1][3] = wmma16(a1, b3, acc[1][3]);
    }

    const int col = lane & 15;
    const int mb  = (lane & 16) ? 8 : 0;
#pragma unroll
    for (int i = 0; i < 2; ++i)
#pragma unroll
        for (int t = 0; t < 4; ++t)
#pragma unroll
            for (int r = 0; r < 8; ++r) {
                const int q = q0 + i * 16 + mb + r;
                const int e = h * HD + t * 16 + col;
                val[((size_t)bb * Ssz + q) * EMB + e] = (_Float16)acc[i][t][r];
            }
}

// ---------------------------------------------------------------------------
// K5: o = values @ Wo^T + bo (f32 out). Wave tile 32x64.
// grid = (8192/32, 512/64), block = 32
// ---------------------------------------------------------------------------
__global__ __launch_bounds__(32) void k_out_proj(
    const _Float16* __restrict__ val, const _Float16* __restrict__ Wo,
    const float* __restrict__ bo, float* __restrict__ out)
{
    const int lane = threadIdx.x;
    const int m0 = blockIdx.x * 32;
    const int n0 = blockIdx.y * 64;
    const int row  = lane & 15;
    const int kofs = (lane & 16) ? 8 : 0;

    v8f acc[2][4] = {};
    for (int kb = 0; kb < EMB / 32; ++kb) {
        const int k0 = kb * 32;
        v16h a0 = frag_f16(val + (size_t)(m0 + row) * EMB + k0 + kofs);
        v16h a1 = frag_f16(val + (size_t)(m0 + 16 + row) * EMB + k0 + kofs);
        v16h b0 = frag_f16(Wo + (size_t)(n0 + row) * EMB + k0 + kofs);
        v16h b1 = frag_f16(Wo + (size_t)(n0 + 16 + row) * EMB + k0 + kofs);
        v16h b2 = frag_f16(Wo + (size_t)(n0 + 32 + row) * EMB + k0 + kofs);
        v16h b3 = frag_f16(Wo + (size_t)(n0 + 48 + row) * EMB + k0 + kofs);
        acc[0][0] = wmma16(a0, b0, acc[0][0]);
        acc[0][1] = wmma16(a0, b1, acc[0][1]);
        acc[0][2] = wmma16(a0, b2, acc[0][2]);
        acc[0][3] = wmma16(a0, b3, acc[0][3]);
        acc[1][0] = wmma16(a1, b0, acc[1][0]);
        acc[1][1] = wmma16(a1, b1, acc[1][1]);
        acc[1][2] = wmma16(a1, b2, acc[1][2]);
        acc[1][3] = wmma16(a1, b3, acc[1][3]);
    }

    const int col = lane & 15;
    const int mb  = (lane & 16) ? 8 : 0;
#pragma unroll
    for (int i = 0; i < 2; ++i)
#pragma unroll
        for (int t = 0; t < 4; ++t)
#pragma unroll
            for (int r = 0; r < 8; ++r) {
                const int n = n0 + t * 16 + col;
                out[(size_t)(m0 + i * 16 + mb + r) * EMB + n] = acc[i][t][r] + bo[n];
            }
}

extern "C" void kernel_launch(void* const* d_in, const int* in_sizes, int n_in,
                              void* d_out, int out_size, void* d_ws, size_t ws_size,
                              hipStream_t stream) {
    const float* x    = (const float*)d_in[0];
    const float* inp  = (const float*)d_in[1];
    const float* Wqkv = (const float*)d_in[2];
    const float* bqkv = (const float*)d_in[3];
    const float* Wo   = (const float*)d_in[4];
    const float* bo   = (const float*)d_in[5];

    float* out  = (float*)d_out;                       // o: [8,1024,512]
    float* attn = out + (size_t)Bsz * Ssz * EMB;       // attention: [8,8,1024,1024]

    // Workspace layout (f16 halves):
    const size_t perT = (size_t)Bsz * NH * Ssz * HD;   // 4,194,304
    _Float16* qws  = (_Float16*)d_ws;
    _Float16* kws  = qws  + perT;
    _Float16* vtws = kws  + perT;
    _Float16* val  = vtws + perT;
    _Float16* wq16 = val  + perT;                      // [1536, 576]
    _Float16* wo16 = wq16 + (size_t)N_QKV * KDIM;      // [512, 512]
    _Float16* xc   = wo16 + (size_t)EMB * EMB;         // [8192, 576]
    // total = 16.78M + 0.88M + 0.26M + 4.72M halves ~= 45.3 MB

    const int nWq8 = (N_QKV * KDIM) / 8;               // 110592
    const int nWo8 = (EMB * EMB) / 8;                  // 32768
    k_cvt16<<<dim3((nWq8 + 255) / 256), 256, 0, stream>>>(Wqkv, wq16, nWq8);
    k_cvt16<<<dim3((nWo8 + 255) / 256), 256, 0, stream>>>(Wo, wo16, nWo8);
    k_concat16<<<dim3(Bsz * Ssz), 128, 0, stream>>>(x, inp, xc);

    k_qkv_gemm<<<dim3((Bsz * Ssz) / 32, N_QKV / 64), 32, 0, stream>>>(
        xc, wq16, bqkv, qws, kws, vtws);
    k_logits<<<dim3(Ssz / 32, Ssz / 64, Bsz * NH), 32, 0, stream>>>(qws, kws, attn);
    k_softmax<<<dim3(Bsz * NH * Ssz), 256, 0, stream>>>(attn);
    k_attn_v<<<dim3(Ssz / 32, Bsz * NH), 32, 0, stream>>>(attn, vtws, val);
    k_out_proj<<<dim3((Bsz * Ssz) / 32, EMB / 64), 32, 0, stream>>>(val, wo16, bo, out);
}
